// MultiHeadGraphTransformerLayer_56169582297732
// MI455X (gfx1250) — matrix operand order repaired and checked by reading, not compiled, not run on hardware
//
#include <hip/hip_runtime.h>
#include <hip/hip_bf16.h>
#include <math.h>

#define DIM 256
#define NH 8
#define HD 32
#define NN 50000
#define EE 800000
#define FFNDIM 1024

typedef __attribute__((ext_vector_type(16))) __bf16    v16bf;
typedef __attribute__((ext_vector_type(8)))  float     v8f;
typedef __attribute__((ext_vector_type(4)))  unsigned  u32x4;

union FragBF16 { u32x4 u[2]; v16bf v; };

// ---------------------------------------------------------------------------
// WMMA GEMM: C[M x Nc] = A[M x Kd] (bf16, row-major) * Bt[Nc x Kd] (bf16,
// row-major = B transposed) + bias, optional exact GELU, output f32 or bf16.
// One wave computes one 16x16 tile; 8 waves (256 threads) per block along M.
// Fragment loads are contiguous 16B per lane, matching CDNA5 WMMA layouts:
//   A 16x32 bf16 : lane<16 -> row=lane,  K = {kb..kb+7, kb+16..kb+23}
//                  lane>=16 -> row=lane-16, K = {kb+8..kb+15, kb+24..kb+31}
//   B 32x16 bf16 : lane<16 -> col=lane,  K = {kb..kb+7, kb+16..kb+23}
//                  lane>=16 -> col=lane-16, K = {kb+8..kb+15, kb+24..kb+31}
//   C 16x16 f32  : vgpr r -> row = r + 8*(lane>=16), col = lane&15
// ---------------------------------------------------------------------------
template <bool GELU, bool OUTBF16>
__global__ __launch_bounds__(256) void wmma_gemm_bf16(
    const __hip_bfloat16* __restrict__ A, const __hip_bfloat16* __restrict__ Bt,
    const float* __restrict__ bias, float* __restrict__ Cf,
    __hip_bfloat16* __restrict__ Cb, int M, int Kd, int Nc) {
  const int lane = threadIdx.x & 31;
  const int wave = threadIdx.x >> 5;
  const int mt = blockIdx.x * 8 + wave;  // wave-uniform
  if (mt * 16 >= M) return;              // full-wave exit, EXEC stays all-1s
  const int nt = blockIdx.y;
  const int m0 = mt * 16;
  const int n0 = nt * 16;
  const int half = lane >> 4;
  const int l15 = lane & 15;

  const __hip_bfloat16* arow = A + (size_t)(m0 + l15) * Kd + half * 8;
  const __hip_bfloat16* brow = Bt + (size_t)(n0 + l15) * Kd + half * 8;

  v8f acc = {};
  for (int kb = 0; kb < Kd; kb += 32) {
    FragBF16 fa, fb;
    fa.u[0] = *(const u32x4*)(arow + kb);
    fa.u[1] = *(const u32x4*)(arow + kb + 16);
    fb.u[0] = *(const u32x4*)(brow + kb);
    fb.u[1] = *(const u32x4*)(brow + kb + 16);
    if (kb + 32 < Kd) {
      __builtin_prefetch(arow + kb + 32, 0, 3);
      __builtin_prefetch(brow + kb + 32, 0, 3);
    }
    acc = __builtin_amdgcn_wmma_f32_16x16x32_bf16(
        /*neg_a=*/false, fa.v, /*neg_b=*/false, fb.v,
        /*c_mod=*/(short)0, acc, /*reuse_a=*/false, /*reuse_b=*/false);
  }

  const int col = n0 + l15;
  const float bs = bias ? bias[col] : 0.0f;
#pragma unroll
  for (int r = 0; r < 8; ++r) {
    const int row = m0 + r + half * 8;
    float v = acc[r] + bs;
    if (GELU) v = 0.5f * v * (1.0f + erff(v * 0.70710678118654752f));
    if (OUTBF16)
      Cb[(size_t)row * Nc + col] = __float2bfloat16(v);
    else
      Cf[(size_t)row * Nc + col] = v;
  }
}

// ---------------------------------------------------------------------------
// Elementwise f32 -> bf16
// ---------------------------------------------------------------------------
__global__ void k_f32_to_bf16(const float* __restrict__ in,
                              __hip_bfloat16* __restrict__ out, int n) {
  int i = blockIdx.x * 256 + threadIdx.x;
  if (i < n) out[i] = __float2bfloat16(in[i]);
}

// W[K x Nc] f32 -> Wt[Nc x K] bf16
__global__ void k_transpose_bf16(const float* __restrict__ in,
                                 __hip_bfloat16* __restrict__ out, int K,
                                 int Nc) {
  int i = blockIdx.x * 256 + threadIdx.x;
  if (i < K * Nc) {
    int k = i / Nc, n = i % Nc;
    out[(size_t)n * K + k] = __float2bfloat16(in[i]);
  }
}

// ---------------------------------------------------------------------------
// Init scratch: attnC = 0, denom = 0, menc = encode(-inf)
// ---------------------------------------------------------------------------
__global__ void k_init(float* __restrict__ attnC, float* __restrict__ denom,
                       unsigned* __restrict__ menc) {
  int i = blockIdx.x * 256 + threadIdx.x;
  if (i < NN * DIM) attnC[i] = 0.0f;
  if (i < NN * NH) {
    denom[i] = 0.0f;
    menc[i] = 0x007FFFFFu;  // ~bits(-inf): monotone encoding of -inf
  }
}

__device__ __forceinline__ unsigned enc_f32(float x) {
  unsigned u = __float_as_uint(x);
  return (u & 0x80000000u) ? ~u : (u | 0x80000000u);
}
__device__ __forceinline__ float dec_f32(unsigned e) {
  unsigned u = (e & 0x80000000u) ? (e & 0x7FFFFFFFu) : ~e;
  return __uint_as_float(u);
}

// ---------------------------------------------------------------------------
// Pass 1: per (edge,head) score = dot(K[src],Q[dst])/sqrt(HD)+bias; seg-max
// ---------------------------------------------------------------------------
__global__ void k_edge_scores(const float* __restrict__ Q,
                              const float* __restrict__ Kf,
                              const float* __restrict__ eb,
                              const int* __restrict__ src,
                              const int* __restrict__ dst,
                              float* __restrict__ score,
                              unsigned* __restrict__ menc) {
  int i = blockIdx.x * 256 + threadIdx.x;
  if (i >= EE * NH) return;
  int e = i >> 3, hh = i & 7;
  int s = src[e], d = dst[e];
  const float4* kp = (const float4*)(Kf + (size_t)s * DIM + hh * HD);
  const float4* qp = (const float4*)(Q + (size_t)d * DIM + hh * HD);
  float acc = 0.0f;
#pragma unroll
  for (int j = 0; j < 8; ++j) {
    float4 a = kp[j], b = qp[j];
    acc += a.x * b.x + a.y * b.y + a.z * b.z + a.w * b.w;
  }
  float sc = acc * 0.17677669529663687f + eb[e];  // 1/sqrt(32)
  score[i] = sc;
  atomicMax(&menc[(size_t)d * NH + hh], enc_f32(sc));
}

// ---------------------------------------------------------------------------
// Pass 2: ex = exp(score - m[dst]); seg-sum into denom (score overwritten)
// ---------------------------------------------------------------------------
__global__ void k_edge_exp(const int* __restrict__ dst,
                           const unsigned* __restrict__ menc,
                           float* __restrict__ score,
                           float* __restrict__ denom) {
  int i = blockIdx.x * 256 + threadIdx.x;
  if (i >= EE * NH) return;
  int e = i >> 3, hh = i & 7;
  int d = dst[e];
  float m = dec_f32(menc[(size_t)d * NH + hh]);
  float ex = __expf(score[i] - m);
  score[i] = ex;
  atomicAdd(&denom[(size_t)d * NH + hh], ex);
}

// ---------------------------------------------------------------------------
// Pass 3: attnC[dst] += (ex/denom[dst]) * V[src]
// ---------------------------------------------------------------------------
__global__ void k_edge_scatter(const int* __restrict__ src,
                               const int* __restrict__ dst,
                               const float* __restrict__ ex,
                               const float* __restrict__ denom,
                               const float* __restrict__ V,
                               float* __restrict__ attnC) {
  int i = blockIdx.x * 256 + threadIdx.x;
  if (i >= EE * NH) return;
  int e = i >> 3, hh = i & 7;
  int s = src[e], d = dst[e];
  float w = ex[i] / denom[(size_t)d * NH + hh];
  const float* vp = V + (size_t)s * DIM + hh * HD;
  float* op = attnC + (size_t)d * DIM + hh * HD;
#pragma unroll
  for (int j = 0; j < HD; ++j) atomicAdd(&op[j], w * vp[j]);
}

// ---------------------------------------------------------------------------
// out = layernorm(x + y); wave32 per row (8 elems/lane), shfl_xor reduce
// ---------------------------------------------------------------------------
__global__ __launch_bounds__(256) void k_add_layernorm(
    const float* __restrict__ x, const float* __restrict__ y,
    const float* __restrict__ g, const float* __restrict__ b,
    float* __restrict__ outf, __hip_bfloat16* __restrict__ outb, int n) {
  int lane = threadIdx.x & 31;
  int row = blockIdx.x * 8 + (threadIdx.x >> 5);
  if (row >= n) return;
  const float* xr = x + (size_t)row * DIM;
  const float* yr = y + (size_t)row * DIM;
  float v[8];
  float s = 0.0f;
#pragma unroll
  for (int j = 0; j < 8; ++j) {
    int c = lane + j * 32;
    v[j] = xr[c] + yr[c];
    s += v[j];
  }
#pragma unroll
  for (int off = 16; off; off >>= 1) s += __shfl_xor(s, off, 32);
  float mu = s * (1.0f / DIM);
  float var = 0.0f;
#pragma unroll
  for (int j = 0; j < 8; ++j) {
    float d = v[j] - mu;
    var += d * d;
  }
#pragma unroll
  for (int off = 16; off; off >>= 1) var += __shfl_xor(var, off, 32);
  float inv = rsqrtf(var * (1.0f / DIM) + 1e-5f);
#pragma unroll
  for (int j = 0; j < 8; ++j) {
    int c = lane + j * 32;
    float o = (v[j] - mu) * inv * g[c] + b[c];
    outf[(size_t)row * DIM + c] = o;
    if (outb) outb[(size_t)row * DIM + c] = __float2bfloat16(o);
  }
}

// ---------------------------------------------------------------------------
extern "C" void kernel_launch(void* const* d_in, const int* in_sizes, int n_in,
                              void* d_out, int out_size, void* d_ws,
                              size_t ws_size, hipStream_t stream) {
  const float* h = (const float*)d_in[0];
  const float* eb = (const float*)d_in[1];
  const int* src = (const int*)d_in[2];
  const int* dst = (const int*)d_in[3];
  const float* Wq = (const float*)d_in[4];
  const float* Wk = (const float*)d_in[5];
  const float* Wv = (const float*)d_in[6];
  const float* Wp = (const float*)d_in[7];
  const float* bp = (const float*)d_in[8];
  const float* ln1_g = (const float*)d_in[9];
  const float* ln1_b = (const float*)d_in[10];
  const float* ln2_g = (const float*)d_in[11];
  const float* ln2_b = (const float*)d_in[12];
  const float* W1 = (const float*)d_in[13];
  const float* b1 = (const float*)d_in[14];
  const float* W2 = (const float*)d_in[15];
  const float* b2 = (const float*)d_in[16];
  float* out = (float*)d_out;

  // --- workspace carve (all offsets 256B aligned) ---
  char* ws = (char*)d_ws;
  size_t off = 0;
  auto alloc = [&](size_t bytes) {
    void* p = ws + off;
    off = (off + bytes + 255) & ~(size_t)255;
    return p;
  };
  __hip_bfloat16* hb = (__hip_bfloat16*)alloc((size_t)NN * DIM * 2);
  __hip_bfloat16* wqt = (__hip_bfloat16*)alloc((size_t)DIM * DIM * 2);
  __hip_bfloat16* wkt = (__hip_bfloat16*)alloc((size_t)DIM * DIM * 2);
  __hip_bfloat16* wvt = (__hip_bfloat16*)alloc((size_t)DIM * DIM * 2);
  __hip_bfloat16* wpt = (__hip_bfloat16*)alloc((size_t)DIM * DIM * 2);
  __hip_bfloat16* w1t = (__hip_bfloat16*)alloc((size_t)DIM * FFNDIM * 2);
  __hip_bfloat16* w2t = (__hip_bfloat16*)alloc((size_t)FFNDIM * DIM * 2);
  float* Qb = (float*)alloc((size_t)NN * DIM * 4);
  float* Kb = (float*)alloc((size_t)NN * DIM * 4);
  float* Vb = (float*)alloc((size_t)NN * DIM * 4);
  float* score = (float*)alloc((size_t)EE * NH * 4);
  unsigned* menc = (unsigned*)alloc((size_t)NN * NH * 4);
  float* denom = (float*)alloc((size_t)NN * NH * 4);
  float* attnC = (float*)alloc((size_t)NN * DIM * 4);
  __hip_bfloat16* attnCb = (__hip_bfloat16*)alloc((size_t)NN * DIM * 2);
  float* h1 = (float*)alloc((size_t)NN * DIM * 4);
  __hip_bfloat16* h1b = (__hip_bfloat16*)alloc((size_t)NN * DIM * 2);
  __hip_bfloat16* a1b = (__hip_bfloat16*)alloc((size_t)NN * FFNDIM * 2);
  float* f2 = (float*)alloc((size_t)NN * DIM * 4);
  float* attn_out = Qb;  // reuse: Q dead after score pass

  const int ND = NN * DIM;                 // 12.8M
  const int gElem = (ND + 255) / 256;      // elementwise grid
  const int gEdge = (EE * NH + 255) / 256; // edge grid
  const int Mtiles = NN / 16;              // 3125 (exact)
  const dim3 blk(256);

  // 0) convert inputs to bf16 (weights transposed for contiguous B frags)
  k_f32_to_bf16<<<gElem, blk, 0, stream>>>(h, hb, ND);
  k_transpose_bf16<<<(DIM * DIM + 255) / 256, blk, 0, stream>>>(Wq, wqt, DIM, DIM);
  k_transpose_bf16<<<(DIM * DIM + 255) / 256, blk, 0, stream>>>(Wk, wkt, DIM, DIM);
  k_transpose_bf16<<<(DIM * DIM + 255) / 256, blk, 0, stream>>>(Wv, wvt, DIM, DIM);
  k_transpose_bf16<<<(DIM * DIM + 255) / 256, blk, 0, stream>>>(Wp, wpt, DIM, DIM);
  k_transpose_bf16<<<(DIM * FFNDIM + 255) / 256, blk, 0, stream>>>(W1, w1t, DIM, FFNDIM);
  k_transpose_bf16<<<(FFNDIM * DIM + 255) / 256, blk, 0, stream>>>(W2, w2t, FFNDIM, DIM);
  k_init<<<gElem, blk, 0, stream>>>(attnC, denom, menc);

  // 1) Q/K/V projections: [N,256] x [256,256]
  const dim3 gProj((Mtiles + 7) / 8, DIM / 16);
  wmma_gemm_bf16<false, false><<<gProj, blk, 0, stream>>>(hb, wqt, nullptr, Qb, nullptr, NN, DIM, DIM);
  wmma_gemm_bf16<false, false><<<gProj, blk, 0, stream>>>(hb, wkt, nullptr, Kb, nullptr, NN, DIM, DIM);
  wmma_gemm_bf16<false, false><<<gProj, blk, 0, stream>>>(hb, wvt, nullptr, Vb, nullptr, NN, DIM, DIM);

  // 2) edge softmax (3 passes, L2-resident N*H state)
  k_edge_scores<<<gEdge, blk, 0, stream>>>(Qb, Kb, eb, src, dst, score, menc);
  k_edge_exp<<<gEdge, blk, 0, stream>>>(dst, menc, score, denom);
  k_edge_scatter<<<gEdge, blk, 0, stream>>>(src, dst, score, denom, Vb, attnC);

  // 3) output projection + residual + LN1
  k_f32_to_bf16<<<gElem, blk, 0, stream>>>(attnC, attnCb, ND);
  wmma_gemm_bf16<false, false><<<gProj, blk, 0, stream>>>(attnCb, wpt, bp, attn_out, nullptr, NN, DIM, DIM);
  k_add_layernorm<<<(NN + 7) / 8, blk, 0, stream>>>(h, attn_out, ln1_g, ln1_b, h1, h1b, NN);

  // 4) FFN: gelu(h1 @ W1 + b1) @ W2 + b2, then residual + LN2 -> out
  const dim3 gF1((Mtiles + 7) / 8, FFNDIM / 16);
  wmma_gemm_bf16<true, true><<<gF1, blk, 0, stream>>>(h1b, w1t, b1, nullptr, a1b, NN, DIM, FFNDIM);
  const dim3 gF2((Mtiles + 7) / 8, DIM / 16);
  wmma_gemm_bf16<false, false><<<gF2, blk, 0, stream>>>(a1b, w2t, b2, f2, nullptr, NN, FFNDIM, DIM);
  k_add_layernorm<<<(NN + 7) / 8, blk, 0, stream>>>(h1, f2, ln2_g, ln2_b, out, nullptr, NN);
}
